// BasicPBC_light_78323023610059
// MI455X (gfx1250) — matrix operand order repaired and checked by reading, not compiled, not run on hardware
//
#include <hip/hip_runtime.h>

typedef __attribute__((ext_vector_type(16))) __bf16 v16bf;
typedef __attribute__((ext_vector_type(8)))  float  v8f;
typedef unsigned short u16;
typedef unsigned int   u32;

#define NN 2048   // keypoints per frame
#define DD 128    // d_model
#define LL 6      // layers
#define HH 4      // heads

union Frag {
    v16bf v;
    u32   u[8];
    u16   s[16];
};

__device__ __forceinline__ u16 f2bf(float f) {
    u32 u = __builtin_bit_cast(u32, f);
    u32 r = u + 0x7fffu + ((u >> 16) & 1u);
    return (u16)(r >> 16);
}

// A-operand permutation within a 32-K chunk: source groups [0,2,1,3]*8 so that a
// lane's 16 A-layout values {half*8..+7, 16+half*8..+7} are contiguous in memory.
__device__ __forceinline__ int permk(int kk) {
    int g = kk >> 3;
    return (((g & 1) << 4) | ((g >> 1) << 3)) + (kk & 7);
}

// knx activation layout: element (c, n) at [(c>>5)*NN + n]*32 + (c&31)
__device__ __forceinline__ size_t knx(int c, int n) {
    return ((size_t)(c >> 5) * NN + n) * 32 + (c & 31);
}

// 16 contiguous bf16 for this lane (half*16 offset applied by caller) -> 8 dwords
__device__ __forceinline__ void loadFrag16(Frag& f, const u16* p) {
#pragma unroll
    for (int j = 0; j < 8; ++j) f.u[j] = *(const u32*)(p + 2 * j);
}

// ---------------------------------------------------------------------------
// bf16 GEMM: C[M,NN] = W[M,K] * X[K,NN] + bias
//   W: [M][K/32][32]   X: knx [K/32][n][32]
//   Each wave: 16 rows x 64 cols (4 accumulators, A reused 4x).
// mode 0: f32 [c,n] (outF) and/or bf16 knx (outB)
// mode 1: bf16 [h][n][32]   (Q: A-operand of scores),  c = d*4+h
// mode 2: bf16 [h*32+dh][m]     (V: B-operand of PV)
// mode 3: bf16 [h][n][32]       (K: B-operand of scores)
// ---------------------------------------------------------------------------
__global__ __launch_bounds__(128) void gemm_bf16_kernel(
    const u16* __restrict__ W, const u16* __restrict__ X,
    const float* __restrict__ bias, int M, int K,
    float* __restrict__ outF, u16* __restrict__ outB, int mode)
{
    const int lane = threadIdx.x & 31, wave = threadIdx.x >> 5;
    const int l16 = lane & 15, half = lane >> 4;
    const int m0 = blockIdx.y * 16;
    const int n0 = (blockIdx.x * 4 + wave) * 64;
    const int KC = K >> 5;

    v8f acc[4] = {};
    for (int kc = 0; kc < KC; ++kc) {
        Frag a;
        loadFrag16(a, W + ((size_t)(m0 + l16) * KC + kc) * 32 + half * 16);
        const u16* Xc = X + ((size_t)kc * NN) * 32 + half * 16;
        if (kc + 1 < KC)
            __builtin_prefetch((const void*)(Xc + ((size_t)NN + n0 + l16) * 32), 0, 0);
#pragma unroll
        for (int t = 0; t < 4; ++t) {
            Frag b;
            loadFrag16(b, Xc + (size_t)(n0 + t * 16 + l16) * 32);
            acc[t] = __builtin_amdgcn_wmma_f32_16x16x32_bf16(
                false, a.v, false, b.v, (short)0, acc[t], false, false);
        }
    }

#pragma unroll
    for (int t = 0; t < 4; ++t) {
#pragma unroll
        for (int v = 0; v < 8; ++v) {
            int r = m0 + v + 8 * half;        // C layout: row = vgpr + 8*half
            int n = n0 + t * 16 + l16;        // C layout: col = lane%16
            float val = acc[t][v] + bias[r];
            if (mode == 0) {
                if (outF) outF[(size_t)r * NN + n] = val;
                if (outB) outB[knx(r, n)] = f2bf(val);
            } else if (mode == 1) {           // Q: [h][n][32]
                int h = r & 3, d = r >> 2;
                outB[((size_t)h * NN + n) * 32 + d] = f2bf(val);
            } else if (mode == 2) {           // V: [h*32+dh][m]
                int h = r & 3, dh = r >> 2;
                outB[(size_t)(h * 32 + dh) * NN + n] = f2bf(val);
            } else {                          // K: [h][n][32]
                int h = r & 3, d = r >> 2;
                outB[((size_t)h * NN + n) * 32 + d] = f2bf(val);
            }
        }
    }
}

// ---------------------------------------------------------------------------
// Flash attention, one head per blockIdx.y, 16 query rows per wave.
// QT: [h][n][32]  KT: [h][m][32]  VH: [h*32+dh][m]  msg out: bf16 knx
// ---------------------------------------------------------------------------
__global__ __launch_bounds__(128) void attn_kernel(
    const u16* __restrict__ QT, const u16* __restrict__ KT,
    const u16* __restrict__ VH, u16* __restrict__ msgB)
{
    __shared__ u16 pbuf[4][16][32];     // per-wave P tile, column m at slot m

    const int lane = threadIdx.x & 31, wave = threadIdx.x >> 5;
    const int l16 = lane & 15, half = lane >> 4;
    const int h  = blockIdx.y;
    const int n0 = (blockIdx.x * 4 + wave) * 16;
    const float rs = 0.17677669529663687f;   // 1/sqrt(32)

    Frag aq;
    loadFrag16(aq, QT + ((size_t)h * NN + n0 + l16) * 32 + half * 16);

    // contiguous LDS column slots for this lane's two score columns
    const int pos0 = l16;                       // m=l16      -> slot m
    const int pos1 = l16 + 16;                  // m=16+l16   -> slot m

    v8f acc0 = {}, acc1 = {};
    float mrow[8], lrow[8];
#pragma unroll
    for (int v = 0; v < 8; ++v) { mrow[v] = -INFINITY; lrow[v] = 0.f; }

    for (int m0 = 0; m0 < NN; m0 += 32) {
        Frag bk0, bk1;
        loadFrag16(bk0, KT + ((size_t)h * NN + m0 + l16) * 32 + half * 16);
        loadFrag16(bk1, KT + ((size_t)h * NN + m0 + 16 + l16) * 32 + half * 16);
        v8f s0 = {}, s1 = {};
        s0 = __builtin_amdgcn_wmma_f32_16x16x32_bf16(false, aq.v, false, bk0.v, (short)0, s0, false, false);
        s1 = __builtin_amdgcn_wmma_f32_16x16x32_bf16(false, aq.v, false, bk1.v, (short)0, s1, false, false);

        // online softmax per score row (row = v + 8*half, spread over 16 lanes)
        float scl[8];
#pragma unroll
        for (int v = 0; v < 8; ++v) {
            float a = s0[v] * rs, b = s1[v] * rs;
            float mx = fmaxf(a, b);
#pragma unroll
            for (int msk = 1; msk < 16; msk <<= 1) mx = fmaxf(mx, __shfl_xor(mx, msk, 32));
            float mnew = fmaxf(mrow[v], mx);
            float p0 = __expf(a - mnew);
            float p1 = __expf(b - mnew);
            float sum = p0 + p1;
#pragma unroll
            for (int msk = 1; msk < 16; msk <<= 1) sum += __shfl_xor(sum, msk, 32);
            scl[v]  = __expf(mrow[v] - mnew);
            lrow[v] = lrow[v] * scl[v] + sum;
            mrow[v] = mnew;
            s0[v] = p0; s1[v] = p1;
        }
#pragma unroll
        for (int v = 0; v < 8; ++v) { acc0[v] *= scl[v]; acc1[v] *= scl[v]; }

        // P (C layout) -> LDS column slots -> contiguous A-fragment reload
#pragma unroll
        for (int v = 0; v < 8; ++v) {
            pbuf[wave][v + 8 * half][pos0] = f2bf(s0[v]);
            pbuf[wave][v + 8 * half][pos1] = f2bf(s1[v]);
        }
        Frag ap;
        loadFrag16(ap, &pbuf[wave][l16][0] + half * 16);

        Frag bv0, bv1;
        loadFrag16(bv0, VH + (size_t)(h * 32 + l16) * NN + m0 + half * 16);
        loadFrag16(bv1, VH + (size_t)(h * 32 + 16 + l16) * NN + m0 + half * 16);
        acc0 = __builtin_amdgcn_wmma_f32_16x16x32_bf16(false, ap.v, false, bv0.v, (short)0, acc0, false, false);
        acc1 = __builtin_amdgcn_wmma_f32_16x16x32_bf16(false, ap.v, false, bv1.v, (short)0, acc1, false, false);
    }

#pragma unroll
    for (int v = 0; v < 8; ++v) {
        float inv = 1.f / lrow[v];
        int n = n0 + v + 8 * half;
        int c0 = l16 * 4 + h;
        int c1 = (16 + l16) * 4 + h;
        msgB[knx(c0, n)] = f2bf(acc0[v] * inv);
        msgB[knx(c1, n)] = f2bf(acc1[v] * inv);
    }
}

// ---------------------------------------------------------------------------
// InstanceNorm1d (affine=False) over N + ReLU, bf16 knx out. One channel/block.
// ---------------------------------------------------------------------------
__global__ __launch_bounds__(256) void inorm_relu_kernel(
    const float* __restrict__ x, u16* __restrict__ y)
{
    const int c = blockIdx.x;
    const float* row = x + (size_t)c * NN;
    float s = 0.f, s2 = 0.f;
    for (int i = threadIdx.x; i < NN; i += 256) { float t = row[i]; s += t; s2 += t * t; }
#pragma unroll
    for (int msk = 1; msk < 32; msk <<= 1) { s += __shfl_xor(s, msk, 32); s2 += __shfl_xor(s2, msk, 32); }
    __shared__ float red[16];
    __shared__ float stat[2];
    int wid = threadIdx.x >> 5;
    if ((threadIdx.x & 31) == 0) { red[wid] = s; red[8 + wid] = s2; }
    __syncthreads();
    if (threadIdx.x == 0) {
        float a = 0.f, b = 0.f;
        for (int i = 0; i < 8; ++i) { a += red[i]; b += red[8 + i]; }
        float mean = a / (float)NN;
        float var = fmaxf(b / (float)NN - mean * mean, 0.f);
        stat[0] = mean; stat[1] = rsqrtf(var + 1e-5f);
    }
    __syncthreads();
    float mean = stat[0], rn = stat[1];
    for (int i = threadIdx.x; i < NN; i += 256)
        y[knx(c, i)] = f2bf(fmaxf((row[i] - mean) * rn, 0.f));
}

// f32 -> bf16, contiguous k within chunk: out [M][K/32][32]
__global__ void cvt_weight_kernel(const float* __restrict__ in, u16* __restrict__ out,
                                  int K, int n) {
    int i = blockIdx.x * blockDim.x + threadIdx.x;
    if (i < n) {
        int m = i / K, k = i % K;
        out[((size_t)m * (K >> 5) + (k >> 5)) * 32 + (k & 31)] = f2bf(in[i]);
    }
}

__global__ void init_desc_kernel(const float* __restrict__ src, float* __restrict__ dst,
                                 u16* __restrict__ topbf, int n) {
    int i = blockIdx.x * blockDim.x + threadIdx.x;
    if (i < n) {
        float v = src[i];
        dst[i] = v;
        topbf[knx(i / NN, i % NN)] = f2bf(v);
    }
}

__global__ void resid_kernel(float* __restrict__ desc, const float* __restrict__ delta,
                             u16* __restrict__ topbf, int n) {
    int i = blockIdx.x * blockDim.x + threadIdx.x;
    if (i < n) {
        float v = desc[i] + delta[i];
        desc[i] = v;
        topbf[knx(i / NN, i % NN)] = f2bf(v);
    }
}

// ---------------------------------------------------------------------------
extern "C" void kernel_launch(void* const* d_in, const int* in_sizes, int n_in,
                              void* d_out, int out_size, void* d_ws, size_t ws_size,
                              hipStream_t stream)
{
    (void)in_sizes; (void)n_in; (void)out_size; (void)ws_size;

    const float* in_desc0 = (const float*)d_in[0];
    const float* in_desc1 = (const float*)d_in[1];
    const float* Wq = (const float*)d_in[2];  const float* bq = (const float*)d_in[3];
    const float* Wk = (const float*)d_in[4];  const float* bk = (const float*)d_in[5];
    const float* Wv = (const float*)d_in[6];  const float* bv = (const float*)d_in[7];
    const float* Wm = (const float*)d_in[8];  const float* bm = (const float*)d_in[9];
    const float* W1 = (const float*)d_in[10]; const float* b1 = (const float*)d_in[11];
    const float* W2 = (const float*)d_in[12]; const float* b2 = (const float*)d_in[13];

    char* p = (char*)d_ws;
    auto carve = [&](size_t bytes) -> char* {
        char* r = p; p += (bytes + 255) & ~(size_t)255; return r;
    };

    u16* wqb = (u16*)carve((size_t)LL * DD * DD * 2);
    u16* wkb = (u16*)carve((size_t)LL * DD * DD * 2);
    u16* wvb = (u16*)carve((size_t)LL * DD * DD * 2);
    u16* wmb = (u16*)carve((size_t)LL * DD * DD * 2);
    u16* w1b = (u16*)carve((size_t)LL * 256 * 256 * 2);
    u16* w2b = (u16*)carve((size_t)LL * DD * 256 * 2);

    u16*   cat[2];   u16* QT[2]; u16* KT[2]; u16* VH[2]; u16* msg[2];
    float* preh[2];  u16* hbf[2]; float* delta[2];
    for (int s = 0; s < 2; ++s) {
        cat[s]   = (u16*)carve((size_t)256 * NN * 2);   // knx chunks 0..3 = bf16 desc
        QT[s]    = (u16*)carve((size_t)DD * NN * 2);
        KT[s]    = (u16*)carve((size_t)DD * NN * 2);
        VH[s]    = (u16*)carve((size_t)DD * NN * 2);
        msg[s]   = (u16*)carve((size_t)DD * NN * 2);
        preh[s]  = (float*)carve((size_t)256 * NN * 4);
        hbf[s]   = (u16*)carve((size_t)256 * NN * 2);
        delta[s] = (float*)carve((size_t)DD * NN * 4);
    }

    auto cvtW = [&](const float* src, u16* dst, int K, int n) {
        cvt_weight_kernel<<<(n + 255) / 256, 256, 0, stream>>>(src, dst, K, n);
    };
    cvtW(Wq, wqb, DD, LL * DD * DD);
    cvtW(Wk, wkb, DD, LL * DD * DD);
    cvtW(Wv, wvb, DD, LL * DD * DD);
    cvtW(Wm, wmb, DD, LL * DD * DD);
    cvtW(W1, w1b, 256, LL * 256 * 256);
    cvtW(W2, w2b, 256, LL * DD * 256);

    float* desc[2] = { (float*)d_out, (float*)d_out + (size_t)DD * NN };
    const int NEl = DD * NN;
    init_desc_kernel<<<(NEl + 255) / 256, 256, 0, stream>>>(in_desc0, desc[0], cat[0], NEl);
    init_desc_kernel<<<(NEl + 255) / 256, 256, 0, stream>>>(in_desc1, desc[1], cat[1], NEl);

    const dim3 g128(NN / 256, DD / 16);     // 16x64 per wave, 4 waves/block
    const dim3 g256(NN / 256, 256 / 16);
    const dim3 gAtt(NN / 64, HH);

    for (int l = 0; l < LL; ++l) {
        const bool cross = (l & 1);
        const u16* wq_l = wqb + (size_t)l * DD * DD;
        const u16* wk_l = wkb + (size_t)l * DD * DD;
        const u16* wv_l = wvb + (size_t)l * DD * DD;
        const u16* wm_l = wmb + (size_t)l * DD * DD;
        const u16* w1_l = w1b + (size_t)l * 256 * 256;
        const u16* w2_l = w2b + (size_t)l * DD * 256;

        for (int s = 0; s < 2; ++s) {
            const int sr = cross ? 1 - s : s;
            gemm_bf16_kernel<<<g128, 128, 0, stream>>>(
                wq_l, cat[s], bq + l * DD, DD, DD, nullptr, QT[s], 1);
            gemm_bf16_kernel<<<g128, 128, 0, stream>>>(
                wk_l, cat[sr], bk + l * DD, DD, DD, nullptr, KT[s], 3);
            gemm_bf16_kernel<<<g128, 128, 0, stream>>>(
                wv_l, cat[sr], bv + l * DD, DD, DD, nullptr, VH[s], 2);

            attn_kernel<<<gAtt, 128, 0, stream>>>(QT[s], KT[s], VH[s], msg[s]);

            // msg' = Wm*msg + bm  -> knx chunks 4..7 of concat buffer
            gemm_bf16_kernel<<<g128, 128, 0, stream>>>(
                wm_l, msg[s], bm + l * DD, DD, DD, nullptr, cat[s] + (size_t)DD * NN, 0);
            // h = W1*[x; msg'] + b1
            gemm_bf16_kernel<<<g256, 128, 0, stream>>>(
                w1_l, cat[s], b1 + l * 256, 256, 256, preh[s], nullptr, 0);
            inorm_relu_kernel<<<256, 256, 0, stream>>>(preh[s], hbf[s]);
            // delta = W2*relu(norm(h)) + b2
            gemm_bf16_kernel<<<g128, 128, 0, stream>>>(
                w2_l, hbf[s], b2 + l * DD, DD, 256, delta[s], nullptr, 0);
        }
        resid_kernel<<<(NEl + 255) / 256, 256, 0, stream>>>(desc[0], delta[0], cat[0], NEl);
        resid_kernel<<<(NEl + 255) / 256, 256, 0, stream>>>(desc[1], delta[1], cat[1], NEl);
    }
}
